// GPT_65343632441585
// MI455X (gfx1250) — matrix operand order repaired and checked by reading, not compiled, not run on hardware
//
#include <hip/hip_runtime.h>
#include <hip/hip_bf16.h>
#include <math.h>

typedef __attribute__((ext_vector_type(16))) __bf16 v16bf;
typedef __attribute__((ext_vector_type(8)))  __bf16 bf16x8;
typedef __attribute__((ext_vector_type(8)))  float  v8f;

#define T_LEN   8192
#define MODEL_D 1024
#define N_HEADS 8
#define ATTN_SCALE 0.1f
#define RMS_EPS 1.1920929e-07f

// ---------------------------------------------------------------------------
// CDNA5 async copy: global -> LDS, 16 bytes per lane, tracked by ASYNCcnt.
// ---------------------------------------------------------------------------
__device__ __forceinline__ void async_b128(uint32_t lds_off, const void* g) {
    asm volatile("global_load_async_to_lds_b128 %0, %1, off"
                 :: "v"(lds_off), "v"((uint64_t)(size_t)g) : "memory");
}
__device__ __forceinline__ void wait_async_0()  {
    asm volatile("s_wait_asynccnt 0x0" ::: "memory");
}
__device__ __forceinline__ void wait_async_12() {
    asm volatile("s_wait_asynccnt 0xc" ::: "memory");
}

// ---------------------------------------------------------------------------
// fp32 -> bf16 bulk convert (8 elements / thread)
// ---------------------------------------------------------------------------
__global__ __launch_bounds__(256) void to_bf16(
    const float* __restrict__ src, __bf16* __restrict__ dst, int n8)
{
    const int i = blockIdx.x * 256 + threadIdx.x;
    if (i >= n8) return;
    const float4* s = (const float4*)src + (size_t)i * 2;
    float4 a = s[0], b = s[1];
    bf16x8 o;
    o[0] = (__bf16)a.x; o[1] = (__bf16)a.y; o[2] = (__bf16)a.z; o[3] = (__bf16)a.w;
    o[4] = (__bf16)b.x; o[5] = (__bf16)b.y; o[6] = (__bf16)b.z; o[7] = (__bf16)b.w;
    *(bf16x8*)(dst + (size_t)i * 8) = o;
}

// ---------------------------------------------------------------------------
// NT GEMM, bf16 in / fp32 out: C[M][N] = scale * A[M][K] @ B[N][K]^T.
// 256x128x64 tiles, 8 waves; wave owns 32 rows x 128 cols (2x8 v8f acc).
// Double-buffered async-to-LDS staging (12 b128 issues / thread / tile).
// Dynamic LDS: A 2x32KB @0 | B 2x16KB @65536  (96 KB)
// ---------------------------------------------------------------------------
__global__ __launch_bounds__(256) void gemm_nt_bf16(
    const __bf16* __restrict__ A, const __bf16* __restrict__ B,
    float* __restrict__ C, int M, int N, int K,
    const float* __restrict__ scale_ptr, int scale_idx)
{
    extern __shared__ __align__(16) char gsm[];
    __bf16* As = (__bf16*)gsm;             // [2][256*64]
    __bf16* Bs = (__bf16*)(gsm + 65536);   // [2][128*64]

    const int tid = threadIdx.x;
    const int w = tid >> 5, L = tid & 31, lh = L & 15, hi = L >> 4;
    const int m0 = blockIdx.y * 256, n0 = blockIdx.x * 128;
    const float scale = scale_ptr[scale_idx];
    const int nk = K >> 6;

    v8f acc[2][8];
    for (int mt = 0; mt < 2; ++mt)
        for (int c = 0; c < 8; ++c)
            for (int r = 0; r < 8; ++r) acc[mt][c][r] = 0.0f;

#define ISSUE_TILE(buf, kt)                                                   \
    {                                                                         \
        const int k0 = (kt) * 64;                                             \
        for (int i = 0; i < 8; ++i) {                                         \
            int cch = tid + i * 256;                                          \
            int row = cch >> 3, o8 = (cch & 7) * 8;                           \
            async_b128((uint32_t)(size_t)(As + (buf) * 16384 + row * 64 + o8),\
                       A + (size_t)(m0 + row) * K + k0 + o8);                 \
        }                                                                     \
        for (int i = 0; i < 4; ++i) {                                         \
            int cch = tid + i * 256;                                          \
            int row = cch >> 3, o8 = (cch & 7) * 8;                           \
            async_b128((uint32_t)(size_t)(Bs + (buf) * 8192 + row * 64 + o8), \
                       B + (size_t)(n0 + row) * K + k0 + o8);                 \
        }                                                                     \
    }

    ISSUE_TILE(0, 0)
    for (int kt = 0; kt < nk; ++kt) {
        const int buf = kt & 1;
        if (kt + 1 < nk) {
            ISSUE_TILE(buf ^ 1, kt + 1)
            wait_async_12();            // tile kt complete (in-order), kt+1 in flight
        } else {
            wait_async_0();
        }
        __syncthreads();

        const __bf16* Ab = As + buf * 16384;
        const __bf16* Bb = Bs + buf * 8192;
        for (int s = 0; s < 2; ++s) {
            v16bf af[2];
            for (int mt = 0; mt < 2; ++mt) {
                const int row = w * 32 + mt * 16 + lh;
                bf16x8 a0 = *(const bf16x8*)(Ab + row * 64 + s * 32 + hi * 8);
                bf16x8 a1 = *(const bf16x8*)(Ab + row * 64 + s * 32 + 16 + hi * 8);
                for (int i = 0; i < 8; ++i) { af[mt][i] = a0[i]; af[mt][i + 8] = a1[i]; }
            }
            for (int c = 0; c < 8; ++c) {
                const int n = c * 16 + lh;
                bf16x8 b0 = *(const bf16x8*)(Bb + n * 64 + s * 32 + hi * 8);
                bf16x8 b1 = *(const bf16x8*)(Bb + n * 64 + s * 32 + 16 + hi * 8);
                v16bf bf;
                for (int i = 0; i < 8; ++i) { bf[i] = b0[i]; bf[i + 8] = b1[i]; }
                acc[0][c] = __builtin_amdgcn_wmma_f32_16x16x32_bf16(
                    false, af[0], false, bf, (short)0, acc[0][c], false, false);
                acc[1][c] = __builtin_amdgcn_wmma_f32_16x16x32_bf16(
                    false, af[1], false, bf, (short)0, acc[1][c], false, false);
            }
        }
        __syncthreads();
    }
#undef ISSUE_TILE

    for (int mt = 0; mt < 2; ++mt)
        for (int c = 0; c < 8; ++c)
            for (int r = 0; r < 8; ++r) {
                int row = m0 + w * 32 + mt * 16 + hi * 8 + r;
                int col = n0 + c * 16 + lh;
                C[(size_t)row * N + col] = scale * acc[mt][c][r];
            }
}

// ---------------------------------------------------------------------------
// Post-process: RMS norm (q,k), rotary, key_offset shift, ve/attn gates.
// One block per token, one wave per head, lane owns 4 consecutive dims.
// ---------------------------------------------------------------------------
__global__ __launch_bounds__(256) void postproc(
    const float* __restrict__ qkv,      // [T][3072]
    const float* __restrict__ x,        // [T][1024]
    const float* __restrict__ ve,       // [T][1024]
    const float* __restrict__ f1,       // [T][128]
    const float* __restrict__ f2,       // [T][128]
    const float* __restrict__ attn_gate_w, // [8][12]
    const float* __restrict__ ve_gate_w,   // [8][12]
    __bf16* __restrict__ qb, __bf16* __restrict__ kb, __bf16* __restrict__ vb,
    float* __restrict__ gates)          // [T][8]
{
    const int t  = blockIdx.x;
    const int h  = threadIdx.x >> 5;
    const int L  = threadIdx.x & 31;
    const int d0 = L * 4;

    const float* qp = qkv + (size_t)t * 3072 + h * 128 + d0;
    float q[4], k[4], v[4];
    for (int i = 0; i < 4; ++i) { q[i] = qp[i]; k[i] = qp[1024 + i]; v[i] = qp[2048 + i]; }

    float sq = q[0]*q[0] + q[1]*q[1] + q[2]*q[2] + q[3]*q[3];
    float sk = k[0]*k[0] + k[1]*k[1] + k[2]*k[2] + k[3]*k[3];
    for (int off = 16; off >= 1; off >>= 1) {
        sq += __shfl_xor(sq, off, 32);
        sk += __shfl_xor(sk, off, 32);
    }
    const float rq = rsqrtf(sq * (1.0f / 128.0f) + RMS_EPS);
    const float rk = rsqrtf(sk * (1.0f / 128.0f) + RMS_EPS);

    float F1[4], F2[4];
    for (int i = 0; i < 4; ++i) {
        F1[i] = f1[(size_t)t * 128 + d0 + i];
        F2[i] = f2[(size_t)t * 128 + d0 + i];
    }
    float qn[4], kn[4];
    for (int i = 0; i < 4; ++i) { qn[i] = q[i] * rq; kn[i] = k[i] * rk; }
    float qs[4] = { qn[1], qn[0], qn[3], qn[2] };
    float ks[4] = { kn[1], kn[0], kn[3], kn[2] };
    float qr[4], kr[4];
    for (int i = 0; i < 4; ++i) {
        qr[i] = F1[i] * qn[i] + F2[i] * qs[i];
        kr[i] = F1[i] * kn[i] + F2[i] * ks[i];
    }

    const size_t base = (size_t)t * 1024 + h * 128 + d0;
    for (int i = 0; i < 4; ++i) qb[base + i] = (__bf16)qr[i];

    if (d0 < 64) {
        for (int i = 0; i < 4; ++i) kb[base + i] = (__bf16)kr[i];
    } else {
        if (t + 1 < T_LEN)
            for (int i = 0; i < 4; ++i)
                kb[(size_t)(t + 1) * 1024 + h * 128 + d0 + i] = (__bf16)kr[i];
        if (t == 0)
            for (int i = 0; i < 4; ++i)
                kb[h * 128 + d0 + i] = (__bf16)kr[i];
    }

    float zv = 0.0f, za = 0.0f;
    for (int j = 0; j < 12; ++j) {
        float xv = x[(size_t)t * 1024 + j];
        zv += xv * ve_gate_w[h * 12 + j];
        za += xv * attn_gate_w[h * 12 + j];
    }
    const float vg = 2.0f / (1.0f + __expf(-zv));
    const float ag = 1.0f / (1.0f + __expf(-za));

    for (int i = 0; i < 4; ++i)
        vb[base + i] = (__bf16)(v[i] + vg * ve[base + i]);
    if (L == 0) gates[t * 8 + h] = ag;
}

// ---------------------------------------------------------------------------
// Windowed varlen attention. Grid (64 q-blocks, 8 heads), 8 waves/block,
// wave w owns query rows [16w,16w+16). 256-key window [qb*128-128, qb*128+128).
// LDS: Ks[256][128] @0 | Vt[128][256] @64K | Vrm/P overlay @128K  (192 KB)
// Out-of-window keys are clamped (their P weights are exactly 0 post-mask).
// ---------------------------------------------------------------------------
__global__ __launch_bounds__(256) void attn_fwd(
    const __bf16* __restrict__ qbuf, const __bf16* __restrict__ kbuf,
    const __bf16* __restrict__ vbuf, const float* __restrict__ gates,
    const int* __restrict__ seqlens, int nseq,
    __bf16* __restrict__ y)             // [T][1024] bf16
{
    extern __shared__ __align__(16) char smem[];
    __bf16* Ks  = (__bf16*)smem;               // [256][128]
    __bf16* Vt  = (__bf16*)(smem + 65536);     // [128][256] (d-major)
    __bf16* Vrm = (__bf16*)(smem + 131072);    // [256][128] staging
    __bf16* Ps  = (__bf16*)(smem + 131072);    // overlay: [8 waves][16][256]

    const int qblk = blockIdx.x;
    const int h    = blockIdx.y;
    const int tid  = threadIdx.x;
    const int w    = tid >> 5;
    const int L    = tid & 31;
    const int lh   = L & 15;
    const int hi   = L >> 4;

    // ---- async stage K and V (row-major); clamp g<0 to 0 (masked later) ----
    {
        const int j = tid;
        int g = qblk * 128 - 128 + j;
        if (g < 0) g = 0;
        const __bf16* ksrc = kbuf + ((size_t)g * 8 + h) * 128;
        const __bf16* vsrc = vbuf + ((size_t)g * 8 + h) * 128;
        for (int it = 0; it < 16; ++it) {
            async_b128((uint32_t)(size_t)(Ks + j * 128 + it * 8), ksrc + it * 8);
            async_b128((uint32_t)(size_t)(Vrm + j * 128 + it * 8), vsrc + it * 8);
        }
    }

    // ---- Q fragments (4 k-steps of 32 over head_dim 128) ----
    v16bf qf[4];
    {
        const int t = qblk * 128 + w * 16 + lh;
        const __bf16* qp = qbuf + ((size_t)t * 8 + h) * 128;
        const int kb8 = hi * 8;
        for (int s = 0; s < 4; ++s) {
            bf16x8 a0 = *(const bf16x8*)(qp + s * 32 + kb8);
            bf16x8 a1 = *(const bf16x8*)(qp + s * 32 + 16 + kb8);
            for (int i = 0; i < 8; ++i) { qf[s][i] = a0[i]; qf[s][i + 8] = a1[i]; }
        }
    }
    wait_async_0();
    __syncthreads();

    // ---- transpose V: Vrm[key][d] -> Vt[d][key] ----
    {
        const int j = tid;
        for (int it = 0; it < 16; ++it) {
            bf16x8 vv = *(const bf16x8*)(Vrm + j * 128 + it * 8);
            for (int z = 0; z < 8; ++z) Vt[(it * 8 + z) * 256 + j] = vv[z];
        }
    }
    __syncthreads();   // Vrm dead from here; region reused as Ps

    // ---- S = Q @ K^T ----
    v8f S[16];
    const int koff = hi * 8;
    for (int c = 0; c < 16; ++c) {
        v8f acc;
        for (int r = 0; r < 8; ++r) acc[r] = 0.0f;
        const int n = c * 16 + lh;
        for (int s = 0; s < 4; ++s) {
            bf16x8 b0 = *(const bf16x8*)(Ks + n * 128 + s * 32 + koff);
            bf16x8 b1 = *(const bf16x8*)(Ks + n * 128 + s * 32 + koff + 16);
            v16bf bf;
            for (int i = 0; i < 8; ++i) { bf[i] = b0[i]; bf[i + 8] = b1[i]; }
            acc = __builtin_amdgcn_wmma_f32_16x16x32_bf16(
                false, qf[s], false, bf, (short)0, acc, false, false);
        }
        S[c] = acc;
    }

    // ---- mask + scale ----
    int gqv[8], segq[8];
    for (int r = 0; r < 8; ++r) {
        int gq = qblk * 128 + w * 16 + hi * 8 + r;
        gqv[r] = gq;
        int s = 0;
        for (int jj = 0; jj < nseq; ++jj) s += (seqlens[jj] <= gq) ? 1 : 0;
        segq[r] = s;
    }
    for (int c = 0; c < 16; ++c) {
        const int gk = qblk * 128 - 128 + c * 16 + lh;
        int segk = 0;
        if (gk >= 0)
            for (int jj = 0; jj < nseq; ++jj) segk += (seqlens[jj] <= gk) ? 1 : 0;
        for (int r = 0; r < 8; ++r) {
            bool ok = (gk >= 0) && (gk <= gqv[r]) && (gqv[r] - gk <= 128) &&
                      (segk == segq[r]);
            S[c][r] = ok ? S[c][r] * ATTN_SCALE : -3.0e38f;
        }
    }

    // ---- softmax per row (row spread over the lane's 16-lane half) ----
    for (int r = 0; r < 8; ++r) {
        float m = S[0][r];
        for (int c = 1; c < 16; ++c) m = fmaxf(m, S[c][r]);
        for (int off = 1; off < 16; off <<= 1) m = fmaxf(m, __shfl_xor(m, off, 32));
        float sum = 0.0f;
        for (int c = 0; c < 16; ++c) {
            float p = __expf(S[c][r] - m);
            S[c][r] = p;
            sum += p;
        }
        for (int off = 1; off < 16; off <<= 1) sum += __shfl_xor(sum, off, 32);
        const float inv = 1.0f / sum;
        for (int c = 0; c < 16; ++c) S[c][r] *= inv;
    }

    // ---- stage P (wave-private region) in A-fragment layout ----
    __bf16* pw = Ps + w * 16 * 256;
    for (int c = 0; c < 16; ++c)
        for (int r = 0; r < 8; ++r)
            pw[(hi * 8 + r) * 256 + c * 16 + lh] = (__bf16)S[c][r];
    __syncthreads();

    // ---- O = P @ V, gate, store bf16 ----
    const int kb8 = hi * 8;
    for (int dt = 0; dt < 8; ++dt) {
        v8f acc;
        for (int r = 0; r < 8; ++r) acc[r] = 0.0f;
        for (int ks = 0; ks < 8; ++ks) {
            bf16x8 a0 = *(const bf16x8*)(pw + lh * 256 + ks * 32 + kb8);
            bf16x8 a1 = *(const bf16x8*)(pw + lh * 256 + ks * 32 + 16 + kb8);
            v16bf af;
            for (int i = 0; i < 8; ++i) { af[i] = a0[i]; af[i + 8] = a1[i]; }
            bf16x8 b0 = *(const bf16x8*)(Vt + (dt * 16 + lh) * 256 + ks * 32 + koff);
            bf16x8 b1 = *(const bf16x8*)(Vt + (dt * 16 + lh) * 256 + ks * 32 + koff + 16);
            v16bf bf;
            for (int i = 0; i < 8; ++i) { bf[i] = b0[i]; bf[i + 8] = b1[i]; }
            acc = __builtin_amdgcn_wmma_f32_16x16x32_bf16(
                false, af, false, bf, (short)0, acc, false, false);
        }
        for (int r = 0; r < 8; ++r) {
            const int t = qblk * 128 + w * 16 + hi * 8 + r;
            const float g = gates[t * 8 + h];
            y[(size_t)t * 1024 + h * 128 + dt * 16 + lh] = (__bf16)(acc[r] * g);
        }
    }
}

// ---------------------------------------------------------------------------
extern "C" void kernel_launch(void* const* d_in, const int* in_sizes, int n_in,
                              void* d_out, int out_size, void* d_ws, size_t ws_size,
                              hipStream_t stream) {
    const float* x          = (const float*)d_in[0];
    const float* qkvo_w     = (const float*)d_in[1];
    const float* ve         = (const float*)d_in[2];
    const float* sa_lambdas = (const float*)d_in[3];
    const float* factor1    = (const float*)d_in[4];
    const float* factor2    = (const float*)d_in[5];
    const float* attn_gw    = (const float*)d_in[6];
    const float* ve_gw      = (const float*)d_in[7];
    const int*   seqlens    = (const int*)d_in[8];
    const int    nseq       = in_sizes[8];
    float* out = (float*)d_out;

    // workspace layout (bytes)
    char* ws = (char*)d_ws;
    float*  qkv   = (float*)(ws);                     // 96 MB fp32 [T][3072]
    __bf16* qb    = (__bf16*)(ws + 100663296);        // 16 MB
    __bf16* kb    = (__bf16*)(ws + 117440512);        // 16 MB
    __bf16* vb    = (__bf16*)(ws + 134217728);        // 16 MB
    float*  gates = (float*)(ws + 150994944);         // 256 KB
    __bf16* yact  = (__bf16*)(ws + 151257088);        // 16 MB bf16 [T][1024]
    __bf16* xb    = (__bf16*)(ws + 168034304);        // 16 MB bf16 [T][1024]
    __bf16* wb    = (__bf16*)(ws + 184811520);        //  8 MB bf16 [4096][1024]

    const dim3 blk(256);

    // 0) pre-convert activations + weights to bf16
    to_bf16<<<dim3((T_LEN * MODEL_D / 8 + 255) / 256), blk, 0, stream>>>(
        x, xb, T_LEN * MODEL_D / 8);
    to_bf16<<<dim3((4 * MODEL_D * MODEL_D / 8 + 255) / 256), blk, 0, stream>>>(
        qkvo_w, wb, 4 * MODEL_D * MODEL_D / 8);

    const int gemm_smem = 98304;
    hipFuncSetAttribute((const void*)gemm_nt_bf16,
                        hipFuncAttributeMaxDynamicSharedMemorySize, gemm_smem);

    // 1) qkv = 0.9 * x @ Wqkv^T
    gemm_nt_bf16<<<dim3(24, 32), blk, gemm_smem, stream>>>(
        xb, wb, qkv, T_LEN, 3 * MODEL_D, MODEL_D, sa_lambdas, 0);

    // 2) rms + rotary + key_offset + gates
    postproc<<<dim3(T_LEN), blk, 0, stream>>>(
        qkv, x, ve, factor1, factor2, attn_gw, ve_gw, qb, kb, vb, gates);

    // 3) windowed attention
    const int attn_smem = 196608;
    hipFuncSetAttribute((const void*)attn_fwd,
                        hipFuncAttributeMaxDynamicSharedMemorySize, attn_smem);
    attn_fwd<<<dim3(64, N_HEADS), blk, attn_smem, stream>>>(
        qb, kb, vb, gates, seqlens, nseq, yact);

    // 4) out = 1.1 * y @ Wo^T  (fp32 result)
    gemm_nt_bf16<<<dim3(8, 32), blk, gemm_smem, stream>>>(
        yact, wb + 3 * MODEL_D * MODEL_D, out, T_LEN, MODEL_D, MODEL_D,
        sa_lambdas, 1);
}
